// KNNGraph_4406636445921
// MI455X (gfx1250) — matrix operand order repaired and checked by reading, not compiled, not run on hardware
//
#include <hip/hip_runtime.h>
#include <stdint.h>

typedef __attribute__((ext_vector_type(16))) _Float16 v16h;
typedef __attribute__((ext_vector_type(8)))  _Float16 v8h;
typedef __attribute__((ext_vector_type(4)))  _Float16 v4h;
typedef __attribute__((ext_vector_type(8)))  float    v8f;
typedef __attribute__((ext_vector_type(4)))  float    v4f;

#define N_PTS 16384
#define DIM   128
#define KNN   16
#define QPB   64      // queries per block = 4 waves * 16
#define CTILE 32      // candidate rows per LDS tile
#define BIGF  3.0e38f

// ---- d_ws layout: sq (64KB) | xhi (4MB f16) | xlo (4MB f16) ----
#define SQ_ELEMS   N_PTS
#define XHI_BYTE   (N_PTS * 4)
#define XLO_BYTE   (XHI_BYTE + N_PTS * DIM * 2)

// ---------------- async global -> LDS helpers (CDNA5) ----------------
__device__ __forceinline__ void async_copy16(void* lds, const void* gsrc) {
  uint32_t l = (uint32_t)(uintptr_t)lds;          // LDS aperture: low 32 bits = LDS byte addr
  asm volatile("global_load_async_to_lds_b128 %0, %1, off"
               :: "v"(l), "v"(gsrc) : "memory");
}
__device__ __forceinline__ void wait_async0() {
#if __has_builtin(__builtin_amdgcn_s_wait_asynccnt)
  __builtin_amdgcn_s_wait_asynccnt(0);
#else
  asm volatile("s_wait_asynccnt 0x0" ::: "memory");
#endif
}

// ---------------- kernel 1: norms + f16 hi/lo decomposition ----------------
__global__ __launch_bounds__(256) void knn_prep(const float* __restrict__ x,
                                                float* __restrict__ sq,
                                                _Float16* __restrict__ xhi,
                                                _Float16* __restrict__ xlo) {
  int t = blockIdx.x * blockDim.x + threadIdx.x;   // 4 threads per row
  int r = t >> 2, quart = t & 3;
  const float4* src = (const float4*)(x + (size_t)r * DIM + quart * 32);
  _Float16* dh = xhi + (size_t)r * DIM + quart * 32;
  _Float16* dl = xlo + (size_t)r * DIM + quart * 32;
  float part = 0.f;
#pragma unroll
  for (int i = 0; i < 8; ++i) {
    float4 v = src[i];
    part += v.x * v.x + v.y * v.y + v.z * v.z + v.w * v.w;
    v4h h, l;
    _Float16 h0 = (_Float16)v.x; h[0] = h0; l[0] = (_Float16)(v.x - (float)h0);
    _Float16 h1 = (_Float16)v.y; h[1] = h1; l[1] = (_Float16)(v.y - (float)h1);
    _Float16 h2 = (_Float16)v.z; h[2] = h2; l[2] = (_Float16)(v.z - (float)h2);
    _Float16 h3 = (_Float16)v.w; h[3] = h3; l[3] = (_Float16)(v.w - (float)h3);
    ((v4h*)dh)[i] = h;
    ((v4h*)dl)[i] = l;
  }
  part += __shfl_xor(part, 1, 32);
  part += __shfl_xor(part, 2, 32);
  if (quart == 0) sq[r] = part;
}

union FragU { v16h v; v8h h[2]; };

// ---------------- kernel 2: fused d2-GEMM + top-k ----------------
__global__ __launch_bounds__(128) void knn_main(const _Float16* __restrict__ xhi,
                                                const _Float16* __restrict__ xlo,
                                                const float* __restrict__ sq,
                                                int* __restrict__ out) {
  // double-buffered candidate tile (pre-converted f16 hi/lo): 2 x 16KB
  __shared__ __align__(16) _Float16 s_chi[2][CTILE * DIM];
  __shared__ __align__(16) _Float16 s_clo[2][CTILE * DIM];

  const int tid  = threadIdx.x;
  const int lane = tid & 31;
  const int wave = tid >> 5;
  const int q    = lane & 15;
  const int hiL  = lane >> 4;
  const int qbase = blockIdx.x * QPB;
  const int qrow  = wave * 16 + q;
  const int selfIdx = qbase + qrow;

  // ---- loop-invariant query (B) fragments straight from global, scaled by -2 ----
  // B 32x16 f16, contiguous k order used here (A matches): lane<16 -> N=lane, K = kb..kb+15 ; lane>=16 -> K = kb+16..kb+31
  v16h mtwo;
#pragma unroll
  for (int e = 0; e < 16; ++e) mtwo[e] = (_Float16)(-2.0f);

  v16h bh[4], bl[4];
#pragma unroll
  for (int kb = 0; kb < 4; ++kb) {
    FragU uh, ul;
    const _Float16* ph = xhi + (size_t)selfIdx * DIM + kb * 32 + hiL * 16;
    const _Float16* pl = xlo + (size_t)selfIdx * DIM + kb * 32 + hiL * 16;
    uh.h[0] = *(const v8h*)ph;  uh.h[1] = *(const v8h*)(ph + 8);
    ul.h[0] = *(const v8h*)pl;  ul.h[1] = *(const v8h*)(pl + 8);
    bh[kb] = uh.v * mtwo;       // -2*q baked in: accumulator becomes d^2 directly
    bl[kb] = ul.v * mtwo;
  }

  const float sqq = sq[selfIdx];

  float bd[KNN]; int bi[KNN];
#pragma unroll
  for (int t = 0; t < KNN; ++t) { bd[t] = BIGF; bi[t] = -1; }

  // staging assignment: 4 threads per candidate row, 64B (hi) + 64B (lo) each
  const int sr = tid >> 2, squart = tid & 3;
  const int selem = squart * 32;                 // f16 element offset within row

  // ---- prologue: async-prefetch tile 0 into buffer 0 ----
  {
    const _Float16* gh = xhi + (size_t)sr * DIM + selem;
    const _Float16* gl = xlo + (size_t)sr * DIM + selem;
    _Float16* dh = &s_chi[0][sr * DIM + selem];
    _Float16* dl = &s_clo[0][sr * DIM + selem];
#pragma unroll
    for (int i = 0; i < 4; ++i) {
      async_copy16(dh + i * 8, gh + i * 8);
      async_copy16(dl + i * 8, gl + i * 8);
    }
  }

  for (int cb = 0; cb < N_PTS; cb += CTILE) {
    const int cur = (cb / CTILE) & 1;
    wait_async0();        // my async writes into buf[cur] have landed
    __syncthreads();      // everyone's writes visible; everyone done reading buf[cur^1]

    // ---- async-prefetch next tile into the other buffer (overlaps WMMA below) ----
    if (cb + CTILE < N_PTS) {
      const _Float16* gh = xhi + (size_t)(cb + CTILE + sr) * DIM + selem;
      const _Float16* gl = xlo + (size_t)(cb + CTILE + sr) * DIM + selem;
      _Float16* dh = &s_chi[cur ^ 1][sr * DIM + selem];
      _Float16* dl = &s_clo[cur ^ 1][sr * DIM + selem];
#pragma unroll
      for (int i = 0; i < 4; ++i) {
        async_copy16(dh + i * 8, gh + i * 8);
        async_copy16(dl + i * 8, gl + i * 8);
      }
    }

    const _Float16* chB = s_chi[cur];
    const _Float16* clB = s_clo[cur];

#pragma unroll
    for (int s = 0; s < 2; ++s) {
      const int mbase = s * 16 + hiL * 8;

      v4f sc0 = *(const v4f*)(sq + cb + mbase);
      v4f sc1 = *(const v4f*)(sq + cb + mbase + 4);
      v8f c;
      c[0] = sqq + sc0[0]; c[1] = sqq + sc0[1]; c[2] = sqq + sc0[2]; c[3] = sqq + sc0[3];
      c[4] = sqq + sc1[0]; c[5] = sqq + sc1[1]; c[6] = sqq + sc1[2]; c[7] = sqq + sc1[3];

      // A 16x32 f16, contiguous k order used here (B matches): lane<16 -> M=lane, K=kb+[0..15]; lane>=16 -> K=kb+[16..31]
      const int arow = s * 16 + q;
#pragma unroll
      for (int kb = 0; kb < 4; ++kb) {
        const int koff = kb * 32 + hiL * 16;
        FragU ah, al;
        const _Float16* pah = chB + arow * DIM + koff;
        const _Float16* pal = clB + arow * DIM + koff;
        ah.h[0] = *(const v8h*)pah;  ah.h[1] = *(const v8h*)(pah + 8);
        al.h[0] = *(const v8h*)pal;  al.h[1] = *(const v8h*)(pal + 8);
        // f16x3: hi*hi + hi*lo + lo*hi (queries carry -2)
        c = __builtin_amdgcn_wmma_f32_16x16x32_f16(false, ah.v, false, bh[kb], (short)0, c, false, false);
        c = __builtin_amdgcn_wmma_f32_16x16x32_f16(false, ah.v, false, bl[kb], (short)0, c, false, false);
        c = __builtin_amdgcn_wmma_f32_16x16x32_f16(false, al.v, false, bh[kb], (short)0, c, false, false);
      }

      // ---- fused streaming top-16 ----
#pragma unroll
      for (int v = 0; v < 8; ++v) {
        int mg = cb + mbase + v;
        float d = c[v];
        if (mg == selfIdx) d = BIGF;
        if (d < bd[KNN - 1]) {
          bd[KNN - 1] = d; bi[KNN - 1] = mg;
#pragma unroll
          for (int t = KNN - 1; t > 0; --t) {
            float d0 = bd[t - 1], d1 = bd[t];
            int   i0 = bi[t - 1], i1 = bi[t];
            bool sw = d1 < d0;
            bd[t - 1] = sw ? d1 : d0;  bd[t] = sw ? d0 : d1;
            bi[t - 1] = sw ? i1 : i0;  bi[t] = sw ? i0 : i1;
          }
        }
      }
    }
  }

  // ---- merge lane pair (l, l+16) sharing a query ----
#pragma unroll
  for (int t = 0; t < KNN; ++t) {
    float pd = __shfl(bd[t], lane ^ 16, 32);
    int   pi = __shfl(bi[t], lane ^ 16, 32);
    if (hiL == 0 && pd < bd[KNN - 1]) {
      bd[KNN - 1] = pd; bi[KNN - 1] = pi;
#pragma unroll
      for (int u = KNN - 1; u > 0; --u) {
        float d0 = bd[u - 1], d1 = bd[u];
        int   i0 = bi[u - 1], i1 = bi[u];
        bool sw = d1 < d0;
        bd[u - 1] = sw ? d1 : d0;  bd[u] = sw ? d0 : d1;
        bi[u - 1] = sw ? i1 : i0;  bi[u] = sw ? i0 : i1;
      }
    }
  }

  if (hiL == 0) {
    int* o = out + (size_t)selfIdx * KNN;
#pragma unroll
    for (int t = 0; t < KNN; ++t) o[t] = bi[t];
  }
}

// ---------------- launcher ----------------
extern "C" void kernel_launch(void* const* d_in, const int* in_sizes, int n_in,
                              void* d_out, int out_size, void* d_ws, size_t ws_size,
                              hipStream_t stream) {
  (void)in_sizes; (void)n_in; (void)out_size; (void)ws_size;
  const float* x = (const float*)d_in[0];
  unsigned char* ws = (unsigned char*)d_ws;
  float*    sq  = (float*)(ws);
  _Float16* xhi = (_Float16*)(ws + XHI_BYTE);
  _Float16* xlo = (_Float16*)(ws + XLO_BYTE);
  int* out = (int*)d_out;

  knn_prep<<<(N_PTS * 4) / 256, 256, 0, stream>>>(x, sq, xhi, xlo);
  knn_main<<<N_PTS / QPB, 128, 0, stream>>>(xhi, xlo, sq, out);
}
